// BahdanauAttentionCoverage_11527692222938
// MI455X (gfx1250) — matrix operand order repaired and checked by reading, not compiled, not run on hardware
//
#include <hip/hip_runtime.h>
#include <hip/hip_bf16.h>

#define BB 64
#define SS 2048
#define DD 512
#define UU 512

typedef __attribute__((ext_vector_type(16))) __bf16        v16bf;
typedef __attribute__((ext_vector_type(8)))  float         v8f;
typedef __attribute__((ext_vector_type(8)))  unsigned int  v8u;

__device__ __forceinline__ unsigned int pack2_bf16(float lo, float hi) {
    // round-to-nearest-even f32 -> bf16, pack two into a dword
    unsigned int ulo = __float_as_uint(lo);
    unsigned int uhi = __float_as_uint(hi);
    ulo = (ulo + 0x7FFFu + ((ulo >> 16) & 1u)) >> 16;
    uhi = (uhi + 0x7FFFu + ((uhi >> 16) & 1u)) & 0xFFFF0000u;
    return (ulo & 0xFFFFu) | uhi;
}

// tanh(x) = 1 - 2/(e^{2x}+1); v_exp + v_rcp (both TRANS, co-execute with XDL)
__device__ __forceinline__ float fast_tanh(float x) {
    float t = __expf(2.0f * x);
    return 1.0f - 2.0f * __builtin_amdgcn_rcpf(t + 1.0f);
}

// ---------------------------------------------------------------------------
// Kernel 1: Wh (D x U, f32 row-major) -> packed bf16 pairs (dword = two
// consecutive N columns). B-fragment loads are 16 bf16 loads per lane,
// repacked into K pairs (lane = N column, VGPR j = K pair).
// ---------------------------------------------------------------------------
__global__ __launch_bounds__(256) void convert_wh_kernel(
        const float* __restrict__ Wh, unsigned int* __restrict__ whb) {
    int i = blockIdx.x * 256 + threadIdx.x;          // pair index, D*U/2 total
    float2 p = ((const float2*)Wh)[i];
    whb[i] = pack2_bf16(p.x, p.y);
}

// ---------------------------------------------------------------------------
// Kernel 2: add_base[b,u] = (dec_hidden @ Ws)[b,u] + bs[u] + bh[u] + bc[u]
// ---------------------------------------------------------------------------
__global__ __launch_bounds__(256) void prep_kernel(
        const float* __restrict__ dec, const float* __restrict__ Ws,
        const float* __restrict__ bs,  const float* __restrict__ bh,
        const float* __restrict__ bc,  float* __restrict__ add_base) {
    __shared__ float sdec[DD];
    const int b = blockIdx.x, t = threadIdx.x;
    sdec[t]       = dec[b * DD + t];
    sdec[t + 256] = dec[b * DD + t + 256];
    __syncthreads();
#pragma unroll
    for (int uo = 0; uo < 2; ++uo) {
        const int u = t + uo * 256;
        float acc = bs[u] + bh[u] + bc[u];
        for (int d = 0; d < DD; ++d) acc += sdec[d] * Ws[d * UU + u];
        add_base[b * UU + u] = acc;
    }
}

// ---------------------------------------------------------------------------
// Kernel 3: fused GEMM + tanh + Vw-reduction -> scores[b,s]
//   score[row] = bv + sum_n Vw[n]*tanh( (enc@Wh)[row,n] + add_base[b,n]
//                                        + prev_cov[row]*Wc[n] )
// 8 waves/block, each wave owns a 16-row M strip. The full 16x512 bf16 A
// strip is converted ONCE into 16 register fragments (128 VGPRs) and reused
// across all 32 N-tiles, so the inner loop is just B-loads + v_wmma.
// ---------------------------------------------------------------------------
__global__ __launch_bounds__(256, 1) void score_kernel(
        const float* __restrict__ enc, const float* __restrict__ prev_cov,
        const float* __restrict__ add_base,
        const unsigned short* __restrict__ whb,     // bf16 Wh, row-major
        const float* __restrict__ Wc, const float* __restrict__ Vw,
        const float* __restrict__ bv, float* __restrict__ scores) {
    const int lane = threadIdx.x & 31;
    const int wave = threadIdx.x >> 5;
    const int hi   = lane >> 4;          // which 16-lane half
    const int l15  = lane & 15;
    const long rowBase = (long)blockIdx.x * 128 + (long)wave * 16;
    const int  b = (int)(rowBase >> 11);               // 2048 rows per batch

    // prev_coverage for this lane's 8 rows (C layout: VGPR r -> M = r + 8*hi)
    float pc[8];
#pragma unroll
    for (int r = 0; r < 8; ++r) pc[r] = prev_cov[rowBase + r + 8 * hi];

    float acc[8];
#pragma unroll
    for (int r = 0; r < 8; ++r) acc[r] = 0.0f;

    // ---- Load & convert the wave's whole A strip (16 rows x K=512) once.
    // A layout (16-bit 16x32): lanes 0-15 / 16-31 both hold row M = l15;
    // VGPR j<4 -> K = hi*8 + 2j, VGPR j>=4 -> K = 16 + hi*8 + 2(j-4).
    const float* arow = enc + (rowBase + l15) * (long)DD;
    v8u afrag[DD / 32];
#pragma unroll
    for (int kk = 0; kk < DD / 32; ++kk) {
#pragma unroll
        for (int j = 0; j < 4; ++j) {
            const float2 p0 = *(const float2*)(arow + kk * 32 + hi * 8 + 2 * j);
            const float2 p1 = *(const float2*)(arow + kk * 32 + 16 + hi * 8 + 2 * j);
            afrag[kk][j]     = pack2_bf16(p0.x, p0.y);
            afrag[kk][j + 4] = pack2_bf16(p1.x, p1.y);
        }
    }

    // B column base for this lane: N = nt*16 + l15, K varies over elements
    const unsigned short* bbase =
        whb + l15;

    for (int nt = 0; nt < UU / 16; ++nt) {
        const int   n    = nt * 16 + l15;
        const float addb = add_base[b * UU + n];
        const float vw   = Vw[n];
        const float wc   = Wc[n];

        v8f c = {};
#pragma unroll
        for (int kk = 0; kk < DD / 32; ++kk) {
            // B fragment (32x16 bf16): VGPR j = packed K pair, same K order as A
            const unsigned short* brow = bbase + (long)kk * 32 * UU + nt * 16;
            v8u bu;
#pragma unroll
            for (int j = 0; j < 8; ++j) {
                const int k0 = (j < 4) ? (hi * 8 + 2 * j) : (16 + hi * 8 + 2 * (j - 4));
                bu[j] = (unsigned int)brow[(long)k0 * UU] | ((unsigned int)brow[(long)(k0 + 1) * UU] << 16);
            }

            c = __builtin_amdgcn_wmma_f32_16x16x32_bf16(
                    false, __builtin_bit_cast(v16bf, afrag[kk]),
                    false, __builtin_bit_cast(v16bf, bu),
                    (short)0, c, false, false);
        }
        // fused epilogue: tanh + Vw weighting, no (B,S,U) intermediate
#pragma unroll
        for (int r = 0; r < 8; ++r) {
            float x = c[r] + addb + pc[r] * wc;
            acc[r] += vw * fast_tanh(x);
        }
    }

    // Row M=r+8*hi is striped across lanes (same half) -> xor-reduce width 16
#pragma unroll
    for (int r = 0; r < 8; ++r) {
        float v = acc[r];
        v += __shfl_xor(v, 1, 16);
        v += __shfl_xor(v, 2, 16);
        v += __shfl_xor(v, 4, 16);
        v += __shfl_xor(v, 8, 16);
        if (l15 == 0) scores[rowBase + r + 8 * hi] = v + bv[0];
    }
}

// ---------------------------------------------------------------------------
// Kernel 4: per-batch masked softmax + attn/coverage outputs + context GEMV
// ---------------------------------------------------------------------------
__global__ __launch_bounds__(512) void softmax_ctx_kernel(
        const float* __restrict__ scores, const float* __restrict__ mask,
        const float* __restrict__ prev_cov, const float* __restrict__ enc,
        float* __restrict__ out_ctx, float* __restrict__ out_attn,
        float* __restrict__ out_cov) {
    __shared__ float sattn[SS];
    __shared__ float sred[16];
    const int b = blockIdx.x, t = threadIdx.x;
    const int lane = t & 31, wv = t >> 5;

    // pass 1: load + max
    float m = -3.4e38f;
    for (int s = t; s < SS; s += 512) {
        float v = scores[b * SS + s];
        sattn[s] = v;
        m = fmaxf(m, v);
    }
#pragma unroll
    for (int o = 16; o; o >>= 1) m = fmaxf(m, __shfl_xor(m, o, 32));
    if (lane == 0) sred[wv] = m;
    __syncthreads();
    float M = sred[0];
#pragma unroll
    for (int i = 1; i < 16; ++i) M = fmaxf(M, sred[i]);
    __syncthreads();

    // pass 2: exp * mask, sum  (softmax*mask renorm == exp*mask / sum)
    float sum = 0.0f;
    for (int s = t; s < SS; s += 512) {
        float w = __expf(sattn[s] - M) * mask[b * SS + s];
        sattn[s] = w;
        sum += w;
    }
#pragma unroll
    for (int o = 16; o; o >>= 1) sum += __shfl_xor(sum, o, 32);
    if (lane == 0) sred[wv] = sum;
    __syncthreads();
    float Z = 0.0f;
#pragma unroll
    for (int i = 0; i < 16; ++i) Z += sred[i];
    const float inv = __builtin_amdgcn_rcpf(Z);

    // pass 3: attn + coverage outputs
    for (int s = t; s < SS; s += 512) {
        float a = sattn[s] * inv;
        sattn[s] = a;
        out_attn[b * SS + s] = a;
        out_cov[b * SS + s]  = a + prev_cov[b * SS + s];
    }
    __syncthreads();

    // pass 4: context[b,d] = sum_s attn[s]*enc[b,s,d]; d-contiguous = coalesced
    const int d = t;  // 512 threads == D
    const float* eb = enc + ((long)b * SS) * DD + d;
    float a0 = 0.f, a1 = 0.f, a2 = 0.f, a3 = 0.f;
    for (int s = 0; s < SS; s += 4) {
        a0 += sattn[s + 0] * eb[(long)(s + 0) * DD];
        a1 += sattn[s + 1] * eb[(long)(s + 1) * DD];
        a2 += sattn[s + 2] * eb[(long)(s + 2) * DD];
        a3 += sattn[s + 3] * eb[(long)(s + 3) * DD];
    }
    out_ctx[b * DD + d] = (a0 + a1) + (a2 + a3);
}

// ---------------------------------------------------------------------------
extern "C" void kernel_launch(void* const* d_in, const int* in_sizes, int n_in,
                              void* d_out, int out_size, void* d_ws, size_t ws_size,
                              hipStream_t stream) {
    const float* dec  = (const float*)d_in[0];
    const float* enc  = (const float*)d_in[1];
    const float* mask = (const float*)d_in[2];
    const float* pcov = (const float*)d_in[3];
    const float* Wh   = (const float*)d_in[4];
    const float* bh   = (const float*)d_in[5];
    const float* Ws   = (const float*)d_in[6];
    const float* bs   = (const float*)d_in[7];
    const float* Wc   = (const float*)d_in[8];
    const float* bc   = (const float*)d_in[9];
    const float* Vw   = (const float*)d_in[10];
    const float* bv   = (const float*)d_in[11];

    float* ws         = (float*)d_ws;
    float* add_base   = ws;                                      // B*U floats
    float* scores     = ws + BB * UU;                            // B*S floats
    unsigned int* whb = (unsigned int*)(ws + BB * UU + BB * SS); // D*U/2 dwords

    float* out_ctx  = (float*)d_out;                // (B,D)
    float* out_attn = out_ctx + BB * DD;            // (B,S)
    float* out_cov  = out_attn + BB * SS;           // (B,S,1)

    hipLaunchKernelGGL(convert_wh_kernel, dim3(DD * UU / 2 / 256), dim3(256), 0,
                       stream, Wh, whb);
    hipLaunchKernelGGL(prep_kernel, dim3(BB), dim3(256), 0, stream,
                       dec, Ws, bs, bh, bc, add_base);
    hipLaunchKernelGGL(score_kernel, dim3(BB * SS / 128), dim3(256), 0, stream,
                       enc, pcov, add_base, (const unsigned short*)whb,
                       Wc, Vw, bv, scores);
    hipLaunchKernelGGL(softmax_ctx_kernel, dim3(BB), dim3(512), 0, stream,
                       scores, mask, pcov, enc, out_ctx, out_attn, out_cov);
}
